// GCNGraphLetModel_89541478187027
// MI455X (gfx1250) — matrix operand order repaired and checked by reading, not compiled, not run on hardware
//
#include <hip/hip_runtime.h>
#include <hip/hip_bf16.h>
#include <math.h>

// ---------------- problem constants ----------------
#define BSZ   8
#define NN    15135
#define FF    64
#define GG    73
#define HH    256
#define LLAY  3
#define HFCC  512
#define CCLS  2
#define EDG   242160

#define MM     (BSZ * NN)      // 121080 rows (batch*nodes)
#define MPAD   121088          // next multiple of 16
#define MTILES (MPAD / 16)     // 7568
#define K1     (FF + GG)       // 137
#define K1PAD  160             // 5 k-tiles of 32
#define K1T    5
#define K2T    8               // 256 / 32
#define NTTOT  (HH / 16)       // 16 column tiles
#define MT_PER_WAVE 4

// ---------------- WMMA types ----------------
typedef __attribute__((ext_vector_type(16))) __bf16 v16bf;
typedef __attribute__((ext_vector_type(8)))  float  v8f;

union ABFrag { v16bf v; uint4 u[2]; };

__device__ __forceinline__ __bf16 f2bf(float f) {
  union { float f; unsigned u; } x; x.f = f;
  unsigned r = x.u + 0x7FFFu + ((x.u >> 16) & 1u);  // RNE
  unsigned short h = (unsigned short)(r >> 16);
  union { unsigned short s; __bf16 b; } y; y.s = h;
  return y.b;
}

__device__ __forceinline__ float eluf(float v) {
  return v > 0.f ? v : expm1f(v);
}

// ---------------- utility ----------------
__global__ void zero_f32_kernel(float* p, long long n) {
  long long i = (long long)blockIdx.x * blockDim.x + threadIdx.x;
  if (i < n) p[i] = 0.f;
}

// ---------------- degree / norm ----------------
__global__ void deg_count_kernel(const long long* __restrict__ col, float* __restrict__ cnt) {
  int e = blockIdx.x * blockDim.x + threadIdx.x;
  if (e < EDG) atomicAdd(&cnt[(int)col[e]], 1.0f);
}

__global__ void deg_fin_kernel(const float* __restrict__ cnt,
                               float* __restrict__ dis, float* __restrict__ dinv) {
  int i = blockIdx.x * blockDim.x + threadIdx.x;
  if (i < NN) {
    float d = cnt[i] + 1.0f;            // +1 self loop
    dis[i]  = rsqrtf(d);
    dinv[i] = 1.0f / d;
  }
}

// expand deg_inv over all (batch, node) rows so GEMM epilogue avoids modulo
__global__ void expand_dinv_kernel(const float* __restrict__ dinv, float* __restrict__ dinvM) {
  long long i = (long long)blockIdx.x * blockDim.x + threadIdx.x;
  if (i < MM) dinvM[i] = dinv[(int)(i % NN)];
}

__global__ void edge_norm_kernel(const long long* __restrict__ row,
                                 const long long* __restrict__ col,
                                 const float* __restrict__ dis,
                                 float* __restrict__ nrm) {
  int e = blockIdx.x * blockDim.x + threadIdx.x;
  if (e < EDG) nrm[e] = dis[(int)row[e]] * dis[(int)col[e]];
}

// ---------------- build [x || graphlets] as padded bf16 (lda = K1PAD) ----------------
__global__ void build_hcat_kernel(const float* __restrict__ x,
                                  const float* __restrict__ pe,
                                  __bf16* __restrict__ A) {
  long long idx = (long long)blockIdx.x * blockDim.x + threadIdx.x;
  long long total = (long long)MPAD * K1PAD;
  if (idx >= total) return;
  int  k = (int)(idx % K1PAD);
  long long m = idx / K1PAD;
  float v = 0.f;
  if (m < MM) {
    int b = (int)(m / NN), n = (int)(m % NN);
    if (k < FF)        v = x[((long long)b * NN + n) * FF + k];
    else if (k < K1)   v = pe[(long long)n * GG + (k - FF)];
  }
  A[idx] = f2bf(v);
}

// ---------------- pack W [Ksrc x 256] f32 -> B-fragment bf16 layout ----------------
// lane l of (kt,nt): col n = nt*16 + (l&15), K = kt*32 + (l>>4)*16 + kk, kk=0..15 contiguous
__global__ void pack_w_kernel(const float* __restrict__ W, __bf16* __restrict__ Bp,
                              int Ksrc, int ktiles) {
  int idx = blockIdx.x * blockDim.x + threadIdx.x;
  int total = ktiles * NTTOT * 32 * 16;
  if (idx >= total) return;
  int kk   = idx & 15;
  int lane = (idx >> 4) & 31;
  int nt   = (idx >> 9) % NTTOT;
  int kt   = idx / (16 * 32 * NTTOT);
  int k = kt * 32 + (lane >> 4) * 16 + kk;
  int n = nt * 16 + (lane & 15);
  float v = (k < Ksrc) ? W[(long long)k * HH + n] : 0.f;
  Bp[idx] = f2bf(v);
}

// ---------------- bf16 WMMA GEMM, B register-resident, fused epilogue ----------------
// wave -> one 16-col n-tile, all K B-fragments in registers; loops over 4 m-tiles.
// Writes Craw = A*W and Cagg = Craw * deg_inv[node] + bias (self loop + bias fused).
template <int KT>
__global__ void __launch_bounds__(256)
gemm_bf16_kernel(const __bf16* __restrict__ A, int lda,
                 const __bf16* __restrict__ Bp,
                 float* __restrict__ Craw, float* __restrict__ Cagg,
                 const float* __restrict__ dinvM, const float* __restrict__ bias) {
  int wave = (blockIdx.x * blockDim.x + threadIdx.x) >> 5;
  int lane = threadIdx.x & 31;
  int nt     = wave & (NTTOT - 1);
  int mchunk = wave / NTTOT;
  int lo = lane & 15, hi = lane >> 4;

  // ---- hoist B panel (KT fragments, 16 cols) into registers ----
  ABFrag b[KT];
  const __bf16* bbase = Bp + ((long long)nt * 32 + lane) * 16;
#pragma unroll
  for (int kt = 0; kt < KT; ++kt) {
    const __bf16* bp = bbase + (long long)kt * (NTTOT * 32 * 16);
    b[kt].u[0] = *(const uint4*)(bp);
    b[kt].u[1] = *(const uint4*)(bp + 8);
  }

  const int ncol = nt * 16 + lo;
  const float biasv = bias[ncol];

#pragma unroll 1
  for (int i = 0; i < MT_PER_WAVE; ++i) {
    int mt = mchunk * MT_PER_WAVE + i;
    if (mt >= MTILES) return;
    int rowA = mt * 16 + lo;
    const __bf16* ap = A + (long long)rowA * lda + hi * 16;

    // speculative prefetch of next m-tile's A row (no runtime guard needed)
    if (i + 1 < MT_PER_WAVE)
      __builtin_prefetch(ap + (long long)16 * lda, 0, 1);

    // batched A loads (one clause), then KT back-to-back WMMAs
    ABFrag a[KT];
#pragma unroll
    for (int kt = 0; kt < KT; ++kt) {
      a[kt].u[0] = *(const uint4*)(ap + kt * 32);
      a[kt].u[1] = *(const uint4*)(ap + kt * 32 + 8);
    }

    v8f acc = {};
#pragma unroll
    for (int kt = 0; kt < KT; ++kt)
      acc = __builtin_amdgcn_wmma_f32_16x16x32_bf16(false, a[kt].v, false, b[kt].v,
                                                    (short)0, acc, false, false);

    // D layout: VGPR r, lanes 0-15 -> M=r, lanes 16-31 -> M=8+r; N = lane&15
    int mbase = mt * 16 + 8 * hi;
    float* crp = Craw + (long long)mbase * HH + ncol;
    float* cap = Cagg + (long long)mbase * HH + ncol;
    if (mt * 16 + 16 <= MM) {           // wave-uniform: full tile, straight-line stores
#pragma unroll
      for (int r = 0; r < 8; ++r) {
        float c = acc[r];
        crp[(long long)r * HH] = c;
        cap[(long long)r * HH] = c * dinvM[mbase + r] + biasv;
      }
    } else {                            // only the final partial tile takes this path
#pragma unroll
      for (int r = 0; r < 8; ++r) {
        int m = mbase + r;
        if (m < MM) {
          float c = acc[r];
          crp[(long long)r * HH] = c;
          cap[(long long)r * HH] = c * dinvM[m] + biasv;
        }
      }
    }
  }
}

// ---------------- edge scatter: agg[b,col] += norm[e] * hw[b,row] ----------------
__global__ void scatter_kernel(const long long* __restrict__ row,
                               const long long* __restrict__ col,
                               const float* __restrict__ nrm,
                               const float* __restrict__ hw,
                               float* __restrict__ agg) {
  long long idx = (long long)blockIdx.x * blockDim.x + threadIdx.x;
  long long total = (long long)BSZ * EDG * (HH / 4);
  if (idx >= total) return;
  int  f4 = (int)(idx % (HH / 4));
  long long t = idx / (HH / 4);
  int  e = (int)(t % EDG);
  int  b = (int)(t / EDG);
  int  r = (int)row[e];
  int  c = (int)col[e];
  float w = nrm[e];
  const float4 v = *(const float4*)(hw + ((long long)b * NN + r) * HH + f4 * 4);
  float* dst = agg + ((long long)b * NN + c) * HH + f4 * 4;
  atomicAdd(dst + 0, w * v.x);
  atomicAdd(dst + 1, w * v.y);
  atomicAdd(dst + 2, w * v.z);
  atomicAdd(dst + 3, w * v.w);
}

// ---------------- fused: ELU, bf16 for next conv, fc_W partial dot into g ----------------
__global__ void __launch_bounds__(256)
post_conv_kernel(const float* __restrict__ agg, __bf16* __restrict__ Anext,
                 const float* __restrict__ fcW, const float* __restrict__ fcb,
                 float* __restrict__ g, int layer, int writeBf) {
  int m = blockIdx.x;            // 0 .. MPAD-1
  int f = threadIdx.x;           // 0 .. 255
  __shared__ float red[256];
  if (m >= MM) {                 // pad rows: keep next-conv A zeroed
    if (writeBf) Anext[(long long)m * HH + f] = f2bf(0.f);
    return;
  }
  float v = agg[(long long)m * HH + f];
  float h = eluf(v);
  if (writeBf) Anext[(long long)m * HH + f] = f2bf(h);
  red[f] = h * fcW[f * LLAY + layer];    // stack(...,-1).reshape order: k*L + l
  __syncthreads();
  for (int s = 128; s > 0; s >>= 1) {
    if (f < s) red[f] += red[f + s];
    __syncthreads();
  }
  if (f == 0) {
    if (layer == 0) g[m] = fcb[0] + red[0];
    else            g[m] += red[0];
  }
}

// ---------------- z = elu(g @ lin1_W + lin1_b), one block per (b,j) ----------------
__global__ void __launch_bounds__(256)
lin1_kernel(const float* __restrict__ g, const float* __restrict__ W,
            const float* __restrict__ b, float* __restrict__ z) {
  int j  = blockIdx.x % HFCC;
  int bb = blockIdx.x / HFCC;
  float s = 0.f;
  for (int n = threadIdx.x; n < NN; n += blockDim.x)
    s += g[(long long)bb * NN + n] * W[(long long)n * HFCC + j];
  __shared__ float red[256];
  red[threadIdx.x] = s;
  __syncthreads();
  for (int st = 128; st > 0; st >>= 1) {
    if (threadIdx.x < st) red[threadIdx.x] += red[threadIdx.x + st];
    __syncthreads();
  }
  if (threadIdx.x == 0) z[bb * HFCC + j] = eluf(red[0] + b[j]);
}

// ---------------- out = log_softmax(z @ lin2_W + lin2_b) ----------------
__global__ void __launch_bounds__(128)
final_kernel(const float* __restrict__ z, const float* __restrict__ W,
             const float* __restrict__ b, float* __restrict__ out) {
  int bb = blockIdx.x;
  int t  = threadIdx.x;
  float s0 = 0.f, s1 = 0.f;
  for (int j = t; j < HFCC; j += blockDim.x) {
    float zv = z[bb * HFCC + j];
    s0 += zv * W[j * CCLS + 0];
    s1 += zv * W[j * CCLS + 1];
  }
  __shared__ float r0[128], r1[128];
  r0[t] = s0; r1[t] = s1;
  __syncthreads();
  for (int st = 64; st > 0; st >>= 1) {
    if (t < st) { r0[t] += r0[t + st]; r1[t] += r1[t + st]; }
    __syncthreads();
  }
  if (t == 0) {
    float o0 = r0[0] + b[0], o1 = r1[0] + b[1];
    float mx = fmaxf(o0, o1);
    float lse = mx + logf(expf(o0 - mx) + expf(o1 - mx));
    out[bb * CCLS + 0] = o0 - lse;
    out[bb * CCLS + 1] = o1 - lse;
  }
}

// ---------------- host orchestration ----------------
static inline int iceil(long long a, int b) { return (int)((a + b - 1) / b); }

extern "C" void kernel_launch(void* const* d_in, const int* in_sizes, int n_in,
                              void* d_out, int out_size, void* d_ws, size_t ws_size,
                              hipStream_t stream) {
  const float*     x      = (const float*)d_in[0];
  const long long* eidx   = (const long long*)d_in[2];
  const float*     pe     = (const float*)d_in[3];
  const float*     W1     = (const float*)d_in[4];
  const float*     b1     = (const float*)d_in[5];
  const float*     W2     = (const float*)d_in[6];
  const float*     b2     = (const float*)d_in[7];
  const float*     W3     = (const float*)d_in[8];
  const float*     b3     = (const float*)d_in[9];
  const float*     fcW    = (const float*)d_in[10];
  const float*     fcb    = (const float*)d_in[11];
  const float*     l1W    = (const float*)d_in[12];
  const float*     l1b    = (const float*)d_in[13];
  const float*     l2W    = (const float*)d_in[14];
  const float*     l2b    = (const float*)d_in[15];
  float* out = (float*)d_out;

  const long long* erow = eidx;        // edge_index[0]
  const long long* ecol = eidx + EDG;  // edge_index[1]

  // ---- workspace carve-up ----
  char* ws = (char*)d_ws;
  size_t off = 0;
  auto carve = [&](size_t bytes) -> char* {
    char* p = ws + off;
    off += (bytes + 255) & ~(size_t)255;
    return p;
  };
  __bf16* Abf  = (__bf16*)carve((size_t)MPAD * HH * sizeof(__bf16));   // activations bf16
  float*  hw   = (float*) carve((size_t)MPAD * HH * sizeof(float));    // GEMM out (raw)
  float*  agg  = (float*) carve((size_t)MPAD * HH * sizeof(float));    // aggregated
  __bf16* W1p  = (__bf16*)carve((size_t)K2T * NTTOT * 32 * 16 * sizeof(__bf16));
  __bf16* W2p  = (__bf16*)carve((size_t)K2T * NTTOT * 32 * 16 * sizeof(__bf16));
  __bf16* W3p  = (__bf16*)carve((size_t)K2T * NTTOT * 32 * 16 * sizeof(__bf16));
  float*  cnt  = (float*) carve((size_t)NN * sizeof(float));
  float*  dis  = (float*) carve((size_t)NN * sizeof(float));
  float*  dinv = (float*) carve((size_t)NN * sizeof(float));
  float*  dinvM= (float*) carve((size_t)MPAD * sizeof(float));
  float*  nrm  = (float*) carve((size_t)EDG * sizeof(float));
  float*  g    = (float*) carve((size_t)MPAD * sizeof(float));
  float*  zbuf = (float*) carve((size_t)BSZ * HFCC * sizeof(float));
  (void)ws_size; (void)in_sizes; (void)n_in; (void)out_size;

  // ---- degree / norm ----
  zero_f32_kernel<<<iceil(NN, 256), 256, 0, stream>>>(cnt, NN);
  deg_count_kernel<<<iceil(EDG, 256), 256, 0, stream>>>(ecol, cnt);
  deg_fin_kernel<<<iceil(NN, 256), 256, 0, stream>>>(cnt, dis, dinv);
  expand_dinv_kernel<<<iceil(MM, 256), 256, 0, stream>>>(dinv, dinvM);
  edge_norm_kernel<<<iceil(EDG, 256), 256, 0, stream>>>(erow, ecol, dis, nrm);

  // ---- inputs -> bf16 ----
  build_hcat_kernel<<<iceil((long long)MPAD * K1PAD, 256), 256, 0, stream>>>(x, pe, Abf);
  pack_w_kernel<<<iceil(K1T * NTTOT * 32 * 16, 256), 256, 0, stream>>>(W1, W1p, K1, K1T);
  pack_w_kernel<<<iceil(K2T * NTTOT * 32 * 16, 256), 256, 0, stream>>>(W2, W2p, HH, K2T);
  pack_w_kernel<<<iceil(K2T * NTTOT * 32 * 16, 256), 256, 0, stream>>>(W3, W3p, HH, K2T);

  const int mchunks    = (MTILES + MT_PER_WAVE - 1) / MT_PER_WAVE;    // 1892
  const int gemmBlocks = iceil((long long)NTTOT * mchunks * 32, 256); // 8 waves/block
  const int scatBlocks = iceil((long long)BSZ * EDG * (HH / 4), 256);

  // ---- conv1 ----
  gemm_bf16_kernel<K1T><<<gemmBlocks, 256, 0, stream>>>(Abf, K1PAD, W1p, hw, agg, dinvM, b1);
  scatter_kernel<<<scatBlocks, 256, 0, stream>>>(erow, ecol, nrm, hw, agg);
  post_conv_kernel<<<MPAD, 256, 0, stream>>>(agg, Abf, fcW, fcb, g, 0, 1);

  // ---- conv2 ----
  gemm_bf16_kernel<K2T><<<gemmBlocks, 256, 0, stream>>>(Abf, HH, W2p, hw, agg, dinvM, b2);
  scatter_kernel<<<scatBlocks, 256, 0, stream>>>(erow, ecol, nrm, hw, agg);
  post_conv_kernel<<<MPAD, 256, 0, stream>>>(agg, Abf, fcW, fcb, g, 1, 1);

  // ---- conv3 (no bf16 write needed) ----
  gemm_bf16_kernel<K2T><<<gemmBlocks, 256, 0, stream>>>(Abf, HH, W3p, hw, agg, dinvM, b3);
  scatter_kernel<<<scatBlocks, 256, 0, stream>>>(erow, ecol, nrm, hw, agg);
  post_conv_kernel<<<MPAD, 256, 0, stream>>>(agg, Abf, fcW, fcb, g, 2, 0);

  // ---- readout ----
  lin1_kernel<<<BSZ * HFCC, 256, 0, stream>>>(g, l1W, l1b, zbuf);
  final_kernel<<<BSZ, 128, 0, stream>>>(zbuf, l2W, l2b, out);
}
